// BitNetMLP_73392401154439
// MI455X (gfx1250) — 8-bit matrix operand order repaired and checked by reading, compiled, not run on hardware
//
#include <hip/hip_runtime.h>
#include <hip/hip_bf16.h>
#include <math.h>

#define BSZ  4096
#define HH   4096
#define II   11008
#define EPSQ 1e-8f
#define NTOTF ((float)((long long)BSZ * (long long)II))

#define LDST 80

typedef __attribute__((ext_vector_type(8))) int v8i;
typedef __attribute__((ext_vector_type(4))) int v4i;
typedef __attribute__((ext_vector_type(2))) int v2i;

__device__ __forceinline__ void async_copy16(const signed char* g, signed char* l) {
    unsigned lds = (unsigned)(size_t)(__attribute__((address_space(3))) signed char*)l;
    unsigned long long ga = (unsigned long long)(size_t)g;
    asm volatile("global_load_async_to_lds_b128 %0, %1, off"
                 :: "v"(lds), "v"(ga) : "memory");
}

#define WAIT_ASYNC(n) asm volatile("s_wait_asynccnt %0" :: "i"(n) : "memory")

__device__ __forceinline__ v8i load_a_frag(const signed char* As, int m0, int lane) {
    const int row = m0 + (lane & 15);
    const int kb  = (lane >> 4) * 8;
    const signed char* p = As + row * LDST + kb;
    v2i d0 = *(const v2i*)(p);
    v2i d1 = *(const v2i*)(p + 16);
    v2i d2 = *(const v2i*)(p + 32);
    v2i d3 = *(const v2i*)(p + 48);
    v8i a;
    a[0] = d0[0]; a[1] = d0[1];
    a[2] = d1[0]; a[3] = d1[1];
    a[4] = d2[0]; a[5] = d2[1];
    a[6] = d3[0]; a[7] = d3[1];
    return a;
}

__device__ __forceinline__ v8i load_b_frag(const signed char* Bs, int n0, int lane) {
    const int n  = n0 + (lane & 15);
    const int kb = (lane >> 4) * 8;
    const signed char* p = Bs + n * LDST + kb;
    v2i d0 = *(const v2i*)(p);
    v2i d1 = *(const v2i*)(p + 16);
    v2i d2 = *(const v2i*)(p + 32);
    v2i d3 = *(const v2i*)(p + 48);
    v8i b;
    b[0] = d0[0]; b[1] = d0[1]; b[2] = d1[0]; b[3] = d1[1];
    b[4] = d2[0]; b[5] = d2[1]; b[6] = d3[0]; b[7] = d3[1];
    return b;
}

__device__ __forceinline__ float silu_f(float v) {
    return v / (1.f + __expf(-v));
}

__global__ void bitnet_init_kernel(float* acc) {
    if (threadIdx.x < 2) acc[threadIdx.x] = 0.f;
}

__global__ __launch_bounds__(256) void f32_to_i8_kernel(
        const float* __restrict__ in, signed char* __restrict__ out, int n4) {
    int i = blockIdx.x * 256 + threadIdx.x;
    if (i >= n4) return;
    float4 v = ((const float4*)in)[i];
    char4 c = make_char4((signed char)(int)rintf(v.x),
                         (signed char)(int)rintf(v.y),
                         (signed char)(int)rintf(v.z),
                         (signed char)(int)rintf(v.w));
    ((char4*)out)[i] = c;
}

__global__ __launch_bounds__(256) void gateup_wmma_kernel(
        const signed char* __restrict__ Aq,
        const signed char* __restrict__ Wg,
        const signed char* __restrict__ Wu,
        int* __restrict__ gate_out,
        int* __restrict__ up_out,
        const float* __restrict__ xs_p,
        const float* __restrict__ wsg_p,
        float* __restrict__ sum_abs_silu) {
    __shared__ __align__(16) signed char As[2][128 * LDST];
    __shared__ __align__(16) signed char Bgs[2][64 * LDST];
    __shared__ __align__(16) signed char Bus[2][64 * LDST];
    __shared__ float red[256];

    const int n_blk = blockIdx.x * 64;
    const int m_blk = blockIdx.y * 128;
    const int tid   = threadIdx.x;
    const int lane  = tid & 31;
    const int wave  = tid >> 5;

    v8i accg[4], accu[4];
    const v8i vz = {0, 0, 0, 0, 0, 0, 0, 0};
#pragma unroll
    for (int f = 0; f < 4; ++f) { accg[f] = vz; accu[f] = vz; }

    const float gs = xs_p[0] * wsg_p[0];

    const int brow = tid >> 2;
    const int boff = (tid & 3) * 16;

    auto issue_tiles = [&](int buf, int k0) {
#pragma unroll
        for (int c = tid; c < 512; c += 256) {
            int row = c >> 2, off = (c & 3) * 16;
            async_copy16(Aq + (size_t)(m_blk + row) * HH + k0 + off,
                         &As[buf][row * LDST + off]);
        }
        size_t g = (size_t)(n_blk + brow) * HH + k0 + boff;
        async_copy16(Wg + g, &Bgs[buf][brow * LDST + boff]);
        async_copy16(Wu + g, &Bus[buf][brow * LDST + boff]);
    };

    issue_tiles(0, 0);
    for (int k0 = 0; k0 < HH; k0 += 64) {
        const int cur   = (k0 >> 6) & 1;
        const int knext = (k0 + 64 < HH) ? (k0 + 64) : 0;
        issue_tiles(cur ^ 1, knext);
        WAIT_ASYNC(4);
        __syncthreads();

        v8i a = load_a_frag(As[cur], wave * 16, lane);
        v8i bfr[4];
#pragma unroll
        for (int f = 0; f < 4; ++f) bfr[f] = load_b_frag(Bgs[cur], f * 16, lane);
#pragma unroll
        for (int f = 0; f < 4; ++f)
            accg[f] = __builtin_amdgcn_wmma_i32_16x16x64_iu8(true, a, true, bfr[f],
                                                             accg[f], false, false);
#pragma unroll
        for (int f = 0; f < 4; ++f) bfr[f] = load_b_frag(Bus[cur], f * 16, lane);
#pragma unroll
        for (int f = 0; f < 4; ++f)
            accu[f] = __builtin_amdgcn_wmma_i32_16x16x64_iu8(true, a, true, bfr[f],
                                                             accu[f], false, false);
        __syncthreads();
    }

    float lsum = 0.f;
    const int mrow0 = m_blk + wave * 16 + ((lane >> 4) << 3);
    const int col0  = n_blk + (lane & 15);
#pragma unroll
    for (int f = 0; f < 4; ++f) {
        const int col = col0 + f * 16;
#pragma unroll
        for (int r = 0; r < 8; ++r) {
            size_t idx = (size_t)(mrow0 + r) * II + col;
            int g = accg[f][r];
            int u = accu[f][r];
            gate_out[idx] = g;
            up_out[idx]   = u;
            lsum += fabsf(silu_f((float)g * gs));
        }
    }
    red[tid] = lsum;
    __syncthreads();
#pragma unroll
    for (int s = 128; s > 0; s >>= 1) {
        if (tid < s) red[tid] += red[tid + s];
        __syncthreads();
    }
    if (tid == 0) atomicAdd(sum_abs_silu, red[0]);
}

__global__ __launch_bounds__(256) void pass2_kernel(
        const int* __restrict__ gate, const int* __restrict__ up,
        int* __restrict__ p_out,
        const float* __restrict__ xs_p, const float* __restrict__ wsg_p,
        const float* __restrict__ wsu_p,
        const float* __restrict__ sum_silu, float* __restrict__ sum_abs_t, int n4) {
    __shared__ float red[256];
    const int tid = threadIdx.x;
    int i = blockIdx.x * 256 + tid;
    float lsum = 0.f;
    if (i < n4) {
        const float gs   = xs_p[0] * wsg_p[0];
        const float us   = xs_p[0] * wsu_p[0];
        const float ga_s = sum_silu[0] / NTOTF + EPSQ;
        const float tsc  = ga_s * us;
        int4 g4 = ((const int4*)gate)[i];
        int4 u4 = ((const int4*)up)[i];
        int4 p4;
        {
            float q = fminf(fmaxf(rintf(silu_f((float)g4.x * gs) / ga_s), -128.f), 127.f);
            p4.x = (int)q * u4.x;
            q = fminf(fmaxf(rintf(silu_f((float)g4.y * gs) / ga_s), -128.f), 127.f);
            p4.y = (int)q * u4.y;
            q = fminf(fmaxf(rintf(silu_f((float)g4.z * gs) / ga_s), -128.f), 127.f);
            p4.z = (int)q * u4.z;
            q = fminf(fmaxf(rintf(silu_f((float)g4.w * gs) / ga_s), -128.f), 127.f);
            p4.w = (int)q * u4.w;
        }
        ((int4*)p_out)[i] = p4;
        lsum = fabsf((float)p4.x * tsc) + fabsf((float)p4.y * tsc) +
               fabsf((float)p4.z * tsc) + fabsf((float)p4.w * tsc);
    }
    red[tid] = lsum;
    __syncthreads();
#pragma unroll
    for (int s = 128; s > 0; s >>= 1) {
        if (tid < s) red[tid] += red[tid + s];
        __syncthreads();
    }
    if (tid == 0) atomicAdd(sum_abs_t, red[0]);
}

__global__ __launch_bounds__(256) void pass3_kernel(
        const int* __restrict__ p, signed char* __restrict__ interq,
        const float* __restrict__ xs_p, const float* __restrict__ wsg_p,
        const float* __restrict__ wsu_p, const float* __restrict__ wsd_p,
        const float* __restrict__ sum_silu, const float* __restrict__ sum_t,
        float* __restrict__ out_scale, int n4) {
    int i = blockIdx.x * 256 + threadIdx.x;
    const float ga_s    = sum_silu[0] / NTOTF + EPSQ;
    const float tsc     = ga_s * (xs_p[0] * wsu_p[0]);
    const float inter_s = sum_t[0] / NTOTF + EPSQ;
    if (i == 0) out_scale[0] = inter_s * wsd_p[0];
    if (i >= n4) return;
    int4 p4 = ((const int4*)p)[i];
    const float r = tsc / inter_s;
    char4 c;
    c.x = (signed char)(int)fminf(fmaxf(rintf((float)p4.x * r), -128.f), 127.f);
    c.y = (signed char)(int)fminf(fmaxf(rintf((float)p4.y * r), -128.f), 127.f);
    c.z = (signed char)(int)fminf(fmaxf(rintf((float)p4.z * r), -128.f), 127.f);
    c.w = (signed char)(int)fminf(fmaxf(rintf((float)p4.w * r), -128.f), 127.f);
    ((char4*)interq)[i] = c;
}

__global__ __launch_bounds__(256) void down_wmma_kernel(
        const signed char* __restrict__ Aq,
        const signed char* __restrict__ Wd,
        float* __restrict__ out) {
    __shared__ __align__(16) signed char As[2][128 * LDST];
    __shared__ __align__(16) signed char Bs[2][64 * LDST];

    const int n_blk = blockIdx.x * 64;
    const int m_blk = blockIdx.y * 128;
    const int tid   = threadIdx.x;
    const int lane  = tid & 31;
    const int wave  = tid >> 5;

    v8i acc[4];
    const v8i vz = {0, 0, 0, 0, 0, 0, 0, 0};
#pragma unroll
    for (int f = 0; f < 4; ++f) acc[f] = vz;

    const int brow = tid >> 2;
    const int boff = (tid & 3) * 16;

    auto issue_tiles = [&](int buf, int k0) {
#pragma unroll
        for (int c = tid; c < 512; c += 256) {
            int row = c >> 2, off = (c & 3) * 16;
            async_copy16(Aq + (size_t)(m_blk + row) * II + k0 + off,
                         &As[buf][row * LDST + off]);
        }
        async_copy16(Wd + (size_t)(n_blk + brow) * II + k0 + boff,
                     &Bs[buf][brow * LDST + boff]);
    };

    issue_tiles(0, 0);
    for (int k0 = 0; k0 < II; k0 += 64) {
        const int cur   = (k0 >> 6) & 1;
        const int knext = (k0 + 64 < II) ? (k0 + 64) : 0;
        issue_tiles(cur ^ 1, knext);
        WAIT_ASYNC(3);
        __syncthreads();

        v8i a = load_a_frag(As[cur], wave * 16, lane);
        v8i bfr[4];
#pragma unroll
        for (int f = 0; f < 4; ++f) bfr[f] = load_b_frag(Bs[cur], f * 16, lane);
#pragma unroll
        for (int f = 0; f < 4; ++f)
            acc[f] = __builtin_amdgcn_wmma_i32_16x16x64_iu8(true, a, true, bfr[f],
                                                            acc[f], false, false);
        __syncthreads();
    }

    const int mrow0 = m_blk + wave * 16 + ((lane >> 4) << 3);
    const int col0  = n_blk + (lane & 15);
#pragma unroll
    for (int f = 0; f < 4; ++f) {
        const int col = col0 + f * 16;
#pragma unroll
        for (int r = 0; r < 8; ++r)
            out[(size_t)(mrow0 + r) * HH + col] = (float)acc[f][r];
    }
}

extern "C" void kernel_launch(void* const* d_in, const int* in_sizes, int n_in,
                              void* d_out, int out_size, void* d_ws, size_t ws_size,
                              hipStream_t stream) {
    (void)in_sizes; (void)n_in; (void)out_size; (void)ws_size;
    const float* x       = (const float*)d_in[0];
    const float* x_scale = (const float*)d_in[1];
    const float* qw_gate = (const float*)d_in[2];
    const float* ws_gate = (const float*)d_in[3];
    const float* qw_up   = (const float*)d_in[4];
    const float* ws_up   = (const float*)d_in[5];
    const float* qw_down = (const float*)d_in[6];
    const float* ws_down = (const float*)d_in[7];
    float* out = (float*)d_out;

    char* w = (char*)d_ws;
    const size_t XQ_SZ = (size_t)BSZ * HH;
    const size_t WQ_SZ = (size_t)II * HH;
    const size_t IQ_SZ = (size_t)BSZ * II;
    signed char* xq      = (signed char*)(w);
    signed char* wqg     = (signed char*)(w + XQ_SZ);
    signed char* wqu     = (signed char*)(w + XQ_SZ + WQ_SZ);
    signed char* wqd     = (signed char*)(w + XQ_SZ + 2 * WQ_SZ);
    signed char* interq  = (signed char*)(w + XQ_SZ + 3 * WQ_SZ);
    int*   gate_i32      = (int*)(w + XQ_SZ + 3 * WQ_SZ + IQ_SZ);
    int*   up_i32        = (int*)(w + XQ_SZ + 3 * WQ_SZ + IQ_SZ + 4 * IQ_SZ);
    float* accum         = (float*)(w + XQ_SZ + 3 * WQ_SZ + IQ_SZ + 8 * IQ_SZ);
    float* sum_abs_silu  = accum;
    float* sum_abs_t     = accum + 1;

    bitnet_init_kernel<<<1, 32, 0, stream>>>(accum);

    {
        int n4x = (int)(XQ_SZ / 4);
        f32_to_i8_kernel<<<n4x / 256, 256, 0, stream>>>(x, xq, n4x);
        int n4w = (int)(WQ_SZ / 4);
        f32_to_i8_kernel<<<n4w / 256, 256, 0, stream>>>(qw_gate, wqg, n4w);
        f32_to_i8_kernel<<<n4w / 256, 256, 0, stream>>>(qw_up,   wqu, n4w);
        f32_to_i8_kernel<<<n4w / 256, 256, 0, stream>>>(qw_down, wqd, n4w);
    }

    {
        dim3 grid(II / 64, BSZ / 128);
        gateup_wmma_kernel<<<grid, 256, 0, stream>>>(
            xq, wqg, wqu, gate_i32, up_i32, x_scale, ws_gate, sum_abs_silu);
    }

    {
        int n4 = (int)(IQ_SZ / 4);
        pass2_kernel<<<(n4 + 255) / 256, 256, 0, stream>>>(
            gate_i32, up_i32, gate_i32  ,
            x_scale, ws_gate, ws_up, sum_abs_silu, sum_abs_t, n4);
    }

    {
        int n4 = (int)(IQ_SZ / 4);
        pass3_kernel<<<(n4 + 255) / 256, 256, 0, stream>>>(
            gate_i32, interq, x_scale, ws_gate, ws_up, ws_down,
            sum_abs_silu, sum_abs_t, out + (size_t)BSZ * HH, n4);
    }

    {
        dim3 grid(HH / 64, BSZ / 128);
        down_wmma_kernel<<<grid, 256, 0, stream>>>(interq, wqd, out);
    }
}
